// LinBertSelfAttention_88510686036518
// MI455X (gfx1250) — matrix operand order repaired and checked by reading, not compiled, not run on hardware
//
#include <hip/hip_runtime.h>
#include <hip/hip_bf16.h>

// Problem constants (match reference)
#define B_   8
#define S_   4096
#define HID_ 1024
#define H_   16
#define D_   64
#define BS_  (B_ * S_)
#define EPS_ 1e-6f

typedef __attribute__((ext_vector_type(16))) __bf16 v16bf;
typedef __attribute__((ext_vector_type(8)))  __bf16 bf16x8;
typedef __attribute__((ext_vector_type(4)))  __bf16 bf16x4;
typedef __attribute__((ext_vector_type(8)))  float  v8f;

// ---------------------------------------------------------------------------
// CDNA5 helpers
// ---------------------------------------------------------------------------
__device__ __forceinline__ v8f wmma_bf16(v16bf a, v16bf b, v8f c) {
    // 8 args: (neg_a, A, neg_b, B, c_mod, C, reuse_a, reuse_b)
    return __builtin_amdgcn_wmma_f32_16x16x32_bf16(
        false, a, false, b, (short)0, c, false, false);
}

// Async global->LDS 16-byte copy (ASYNCcnt tracked). The LDS destination
// address is the low 32 bits of the generic __shared__ pointer
__device__ __forceinline__ void async_copy_b128(const void* gptr, void* lptr) {
    unsigned           lds = (unsigned)(uintptr_t)lptr;
    unsigned long long ga  = (unsigned long long)(uintptr_t)gptr;
    asm volatile("global_load_async_to_lds_b128 %0, %1, off"
                 :: "v"(lds), "v"(ga) : "memory");
}
__device__ __forceinline__ void wait_async0() {
    asm volatile("s_wait_asynccnt 0" ::: "memory");
}
// Wait until only the 4 most recent async copies are outstanding (ASYNCcnt
// completes in issue order, so all older copies have landed in LDS).
__device__ __forceinline__ void wait_async4() {
    asm volatile("s_wait_asynccnt 4" ::: "memory");
}

// Branchless phi: elu(x)+1 == (x>0 ? x+1 : exp(x))
__device__ __forceinline__ float phi_fn(float x) {
    float e = __expf(fminf(x, 0.f));   // unconditional exp -> no exec branching
    return x > 0.f ? x + 1.f : e;      // lowers to v_cndmask
}

// A-matrix (16x32 bf16) fragment from LDS tile stored [row][k] (k contiguous).
// Layout: lanes0-15 hold K=0..15 (VGPR i -> K=2i,2i+1), lanes16-31 K=16..31.
__device__ __forceinline__ v16bf load_fragA(const __bf16* base, int stride,
                                            int row, bool hi) {
    const __bf16* p = base + row * stride;
    v16bf a;
#pragma unroll
    for (int vg = 0; vg < 8; ++vg) {
        int kk = 2 * vg + (hi ? 16 : 0);
        a[2 * vg]     = p[kk];
        a[2 * vg + 1] = p[kk + 1];
    }
    return a;
}

// B-matrix (32x16 bf16) fragment from LDS tile stored [n][k] (k contiguous).
// Layout (same k order as A): lanes0-15 hold K=0..15 (VGPR i -> K=2i,2i+1), lanes16-31 K=16..31.
__device__ __forceinline__ v16bf load_fragB(const __bf16* base, int stride,
                                            int col, bool hi) {
    const __bf16* p = base + col * stride;
    v16bf b;
#pragma unroll
    for (int vg = 0; vg < 8; ++vg) {
        int kk = 2 * vg + (hi ? 16 : 0);
        b[2 * vg]     = p[kk];
        b[2 * vg + 1] = p[kk + 1];
    }
    return b;
}

// ---------------------------------------------------------------------------
// Kernel 0: fp32 -> bf16 conversion (weights [n][k] kept as-is; also hidden).
// ---------------------------------------------------------------------------
__global__ __launch_bounds__(256) void cvt_bf16_kernel(const float* __restrict__ src,
                                                       __bf16* __restrict__ dst) {
    size_t i = ((size_t)blockIdx.x * 256 + threadIdx.x) * 4;
    float4 f = *(const float4*)(src + i);
    bf16x4 o;
    o.x = (__bf16)f.x; o.y = (__bf16)f.y; o.z = (__bf16)f.z; o.w = (__bf16)f.w;
    *(bf16x4*)(dst + i) = o;
}

// ---------------------------------------------------------------------------
// Kernel 1: fused QKV projection + phi epilogue.
// grid = (HID/128, BS/128, 3), block = 256 (8 waves).
// Double-buffered LDS; async DMA for tile k+1 overlaps WMMA on tile k.
// ---------------------------------------------------------------------------
#define LT1 40  // LDS row stride in bf16 (80B, multiple of 16B)
#define KSTEPS (HID_ / 32)

__device__ __forceinline__ void stage_tiles(const __bf16* __restrict__ hidB,
                                            const __bf16* __restrict__ W,
                                            __bf16* la, __bf16* lb,
                                            int mb, int nb, int k0, int t) {
#pragma unroll
    for (int i = 0; i < 2; ++i) {
        int lin = t + i * 256;           // 0..511 16-byte chunks
        int r = lin >> 2, c8 = lin & 3;
        async_copy_b128(hidB + (size_t)(mb + r) * HID_ + k0 + c8 * 8,
                        la + r * LT1 + c8 * 8);
        async_copy_b128(W + (size_t)(nb + r) * HID_ + k0 + c8 * 8,
                        lb + r * LT1 + c8 * 8);
    }
}

__global__ __launch_bounds__(256) void qkv_phi_kernel(
    const __bf16* __restrict__ hidB, const float* __restrict__ mask,
    const __bf16* __restrict__ wq, const __bf16* __restrict__ wk,
    const __bf16* __restrict__ wvp,
    const float* __restrict__ bq, const float* __restrict__ bk,
    const float* __restrict__ bvv,
    __bf16* __restrict__ outq, __bf16* __restrict__ outk,
    __bf16* __restrict__ outv) {
    __shared__ __bf16 ldsA[2][128 * LT1];
    __shared__ __bf16 ldsB[2][128 * LT1];

    const int proj = blockIdx.z;
    const __bf16* W    = proj == 0 ? wq   : (proj == 1 ? wk   : wvp);
    const float*  bias = proj == 0 ? bq   : (proj == 1 ? bk   : bvv);
    __bf16*       out  = proj == 0 ? outq : (proj == 1 ? outk : outv);

    const int nb = blockIdx.x * 128;
    const int mb = blockIdx.y * 128;
    const int t = threadIdx.x, lane = t & 31, wid = t >> 5;
    const bool hi = lane >= 16;
    const int ln = lane & 15;
    const int mt0 = (wid & 3) * 2;   // 2 M-tiles per wave
    const int nt0 = (wid >> 2) * 4;  // 4 N-tiles per wave

    v8f acc[2][4] = {};

    // prologue: stage tile 0
    stage_tiles(hidB, W, ldsA[0], ldsB[0], mb, nb, 0, t);

    for (int step = 0; step < KSTEPS; ++step) {
        const int cur = step & 1;
        __syncthreads();   // all waves done reading buffer cur^1 (prev step)
        if (step + 1 < KSTEPS) {
            stage_tiles(hidB, W, ldsA[cur ^ 1], ldsB[cur ^ 1],
                        mb, nb, (step + 1) * 32, t);
            wait_async4();   // newest 4 = next tile; current tile complete
        } else {
            wait_async0();
        }
        __syncthreads();     // current tile visible to all waves

        v16bf afr[2], bfr[4];
#pragma unroll
        for (int mi = 0; mi < 2; ++mi)
            afr[mi] = load_fragA(ldsA[cur], LT1, (mt0 + mi) * 16 + ln, hi);
#pragma unroll
        for (int ni = 0; ni < 4; ++ni)
            bfr[ni] = load_fragB(ldsB[cur], LT1, (nt0 + ni) * 16 + ln, hi);
#pragma unroll
        for (int mi = 0; mi < 2; ++mi)
#pragma unroll
            for (int ni = 0; ni < 4; ++ni)
                acc[mi][ni] = wmma_bf16(afr[mi], bfr[ni], acc[mi][ni]);
    }

    // epilogue: bias + phi (+mask for K), store bf16
    float bvals[4];
#pragma unroll
    for (int ni = 0; ni < 4; ++ni)
        bvals[ni] = bias[nb + (nt0 + ni) * 16 + ln];

#pragma unroll
    for (int mi = 0; mi < 2; ++mi) {
#pragma unroll
        for (int i = 0; i < 8; ++i) {
            const int rowg = mb + (mt0 + mi) * 16 + i + (hi ? 8 : 0);
            float m = 1.f;
            if (proj == 1) m = mask[rowg];      // uniform branch, 1 load/row
#pragma unroll
            for (int ni = 0; ni < 4; ++ni) {
                const int colg = nb + (nt0 + ni) * 16 + ln;
                float val = acc[mi][ni][i] + bvals[ni];
                if (proj != 2) val = phi_fn(val);
                if (proj == 1) val *= m;
                out[(size_t)rowg * HID_ + colg] = (__bf16)val;
            }
        }
    }
}

// ---------------------------------------------------------------------------
// Kernel 2: kv = phi_k^T @ v  (64x64, K=S per head) and z = sum_s phi_k.
// grid = (B*H), block = 256. Register-double-buffered global loads; LDS chunks
// staged transposed: [d][s], [e][s].
// ---------------------------------------------------------------------------
#define LT2 40

__global__ __launch_bounds__(256) void kvz_kernel(
    const __bf16* __restrict__ phiK, const __bf16* __restrict__ vmat,
    float* __restrict__ kvout, float* __restrict__ zout) {
    __shared__ __bf16 ldsK[64 * LT2];
    __shared__ __bf16 ldsV[64 * LT2];
    __shared__ float  zbuf[256];

    const int bh = blockIdx.x;
    const int b = bh >> 4, h = bh & 15;
    const int t = threadIdx.x, lane = t & 31, wid = t >> 5;
    const bool hi = lane >= 16;
    const int ln = lane & 15;
    const int srow = t >> 3, c8 = t & 7;
    const size_t rowbase = (size_t)b * S_;

    const __bf16* pk = phiK + (rowbase + srow) * HID_ + h * 64 + c8 * 8;
    const __bf16* pv = vmat + (rowbase + srow) * HID_ + h * 64 + c8 * 8;

    v8f acc[2] = {};
    float zp = 0.f;

    bf16x8 kk = *(const bf16x8*)pk;
    bf16x8 vv = *(const bf16x8*)pv;

    for (int sc = 0; sc < S_; sc += 32) {
        __syncthreads();
        // transpose-stage 32 s-rows x 64 d/e cols into [d][s] / [e][s]
#pragma unroll
        for (int j = 0; j < 8; ++j) {
            ldsK[(c8 * 8 + j) * LT2 + srow] = kk[j];
            ldsV[(c8 * 8 + j) * LT2 + srow] = vv[j];
        }
        // issue next chunk's loads; the waitcnt lands after this chunk's WMMAs
        bf16x8 kn = kk, vn = vv;
        if (sc + 32 < S_) {
            kn = *(const bf16x8*)(pk + (size_t)(sc + 32) * HID_);
            vn = *(const bf16x8*)(pv + (size_t)(sc + 32) * HID_);
        }
        __syncthreads();
        // z partial sums (thread t owns d = t%64, s-quarter t/64)
        {
            const int d = t & 63, q = t >> 6;
#pragma unroll
            for (int j = 0; j < 8; ++j) zp += (float)ldsK[d * LT2 + q * 8 + j];
        }
        // 16 output tiles (4x4 of 16x16); wave handles 2
#pragma unroll
        for (int u = 0; u < 2; ++u) {
            const int ti = wid * 2 + u, mt = ti >> 2, nt = ti & 3;
            v16bf a  = load_fragA(ldsK, LT2, mt * 16 + ln, hi);
            v16bf bb = load_fragB(ldsV, LT2, nt * 16 + ln, hi);
            acc[u] = wmma_bf16(a, bb, acc[u]);
        }
        kk = kn; vv = vn;
    }

#pragma unroll
    for (int u = 0; u < 2; ++u) {
        const int ti = wid * 2 + u, mt = ti >> 2, nt = ti & 3;
#pragma unroll
        for (int i = 0; i < 8; ++i) {
            int d = mt * 16 + i + (hi ? 8 : 0);
            int e = nt * 16 + ln;
            kvout[(size_t)bh * (D_ * D_) + d * D_ + e] = acc[u][i];
        }
    }
    zbuf[t] = zp;
    __syncthreads();
    if (t < 64)
        zout[bh * D_ + t] = zbuf[t] + zbuf[t + 64] + zbuf[t + 128] + zbuf[t + 192];
}

// ---------------------------------------------------------------------------
// Kernel 3: out = (phi_q @ kv) / (phi_q . z + eps).
// grid = (S/128, B*H), block = 256.  8 waves: wave w = M-tile w, 4 N-tiles.
// phi_q tile staged with async DMA; kv converted fp32->bf16 transposed.
// ---------------------------------------------------------------------------
#define LT3 72  // 144B row stride, multiple of 16B

__global__ __launch_bounds__(256) void attn_out_kernel(
    const __bf16* __restrict__ phiQ, const float* __restrict__ kvin,
    const float* __restrict__ zin, float* __restrict__ dout) {
    __shared__ __bf16 ldsQ[128 * LT3];   // [m][d]
    __shared__ __bf16 ldsKV[64 * LT3];   // [e][d]  (B-form of kv)
    __shared__ float  zs[64];
    __shared__ float  denBuf[128];

    const int sc = blockIdx.x * 128;
    const int bh = blockIdx.y;
    const int b = bh >> 4, h = bh & 15;
    const int t = threadIdx.x, lane = t & 31, wid = t >> 5;
    const bool hi = lane >= 16;
    const int ln = lane & 15;
    const size_t rowbase = (size_t)b * S_ + sc;

    // stage phi_q tile (128 x 64 bf16) via async DMA
#pragma unroll
    for (int i = 0; i < 4; ++i) {
        int lin = t + i * 256;           // 0..1023 16-byte chunks
        int row = lin >> 3, c8 = lin & 7;
        async_copy_b128(phiQ + (rowbase + row) * HID_ + h * 64 + c8 * 8,
                        &ldsQ[row * LT3 + c8 * 8]);
    }

    // stage kv transposed -> bf16 [e][d]
#pragma unroll
    for (int i = 0; i < 16; ++i) {
        int lin = t * 16 + i;            // 0..4095
        int d = lin >> 6, e = lin & 63;
        ldsKV[e * LT3 + d] = (__bf16)kvin[(size_t)bh * (D_ * D_) + lin];
    }
    if (t < 64) zs[t] = zin[bh * D_ + t];

    wait_async0();
    __syncthreads();

    // denominator: one row per thread (t < 128)
    if (t < 128) {
        float s = 0.f;
#pragma unroll
        for (int d = 0; d < 64; ++d) s += (float)ldsQ[t * LT3 + d] * zs[d];
        denBuf[t] = s;
    }

    // numerator via WMMA
    v8f acc[4] = {};
#pragma unroll
    for (int ks = 0; ks < 2; ++ks) {
        v16bf a = load_fragA(ldsQ + ks * 32, LT3, wid * 16 + ln, hi);
#pragma unroll
        for (int nt = 0; nt < 4; ++nt) {
            v16bf bb = load_fragB(ldsKV + ks * 32, LT3, nt * 16 + ln, hi);
            acc[nt] = wmma_bf16(a, bb, acc[nt]);
        }
    }
    __syncthreads();  // denBuf ready for all waves

#pragma unroll
    for (int nt = 0; nt < 4; ++nt) {
        const int colg = h * 64 + nt * 16 + ln;
#pragma unroll
        for (int i = 0; i < 8; ++i) {
            int r = wid * 16 + i + (hi ? 8 : 0);
            float val = acc[nt][i] / (denBuf[r] + EPS_);
            dout[(rowbase + r) * HID_ + colg] = val;
        }
    }
}

// ---------------------------------------------------------------------------
// Launch
// ---------------------------------------------------------------------------
extern "C" void kernel_launch(void* const* d_in, const int* in_sizes, int n_in,
                              void* d_out, int out_size, void* d_ws, size_t ws_size,
                              hipStream_t stream) {
    const float* hidden = (const float*)d_in[0];
    const float* mask   = (const float*)d_in[1];
    const float* Wq     = (const float*)d_in[2];
    const float* bq     = (const float*)d_in[3];
    const float* Wk     = (const float*)d_in[4];
    const float* bk     = (const float*)d_in[5];
    const float* Wv     = (const float*)d_in[6];
    const float* bv     = (const float*)d_in[7];
    float* out = (float*)d_out;

    // Workspace layout
    __bf16* wqb  = (__bf16*)d_ws;
    __bf16* wkb  = wqb + (size_t)HID_ * HID_;
    __bf16* wvb  = wkb + (size_t)HID_ * HID_;
    __bf16* hidB = wvb + (size_t)HID_ * HID_;
    __bf16* phiQ = hidB + (size_t)BS_ * HID_;
    __bf16* phiK = phiQ + (size_t)BS_ * HID_;
    __bf16* vmat = phiK + (size_t)BS_ * HID_;
    float*  kv   = (float*)(vmat + (size_t)BS_ * HID_);
    float*  z    = kv + (size_t)B_ * H_ * D_ * D_;

    // 0) fp32 -> bf16 conversions (weights stay [n][k]; hidden is L2-resident
    //    at 64MB bf16, so the 24 tile-passes in kernel 1 re-read it from L2)
    const int wBlocks = (HID_ * HID_) / (256 * 4);
    cvt_bf16_kernel<<<wBlocks, 256, 0, stream>>>(Wq, wqb);
    cvt_bf16_kernel<<<wBlocks, 256, 0, stream>>>(Wk, wkb);
    cvt_bf16_kernel<<<wBlocks, 256, 0, stream>>>(Wv, wvb);
    const int hBlocks = ((size_t)BS_ * HID_) / (256 * 4);
    cvt_bf16_kernel<<<hBlocks, 256, 0, stream>>>(hidden, hidB);

    // 1) fused QKV projection + phi
    qkv_phi_kernel<<<dim3(HID_ / 128, BS_ / 128, 3), 256, 0, stream>>>(
        hidB, mask, wqb, wkb, wvb, bq, bk, bv, phiQ, phiK, vmat);

    // 2) per-head kv and z
    kvz_kernel<<<dim3(B_ * H_), 256, 0, stream>>>(phiK, vmat, kv, z);

    // 3) output
    attn_out_kernel<<<dim3(S_ / 128, B_ * H_), 256, 0, stream>>>(phiQ, kv, z, out);
}
